// SGNN_29927332119077
// MI455X (gfx1250) — matrix operand order repaired and checked by reading, not compiled, not run on hardware
//
#include <hip/hip_runtime.h>

typedef __attribute__((ext_vector_type(16))) __bf16 v16bf;
typedef __attribute__((ext_vector_type(8)))  __bf16 v8bf;
typedef __attribute__((ext_vector_type(8)))  float  v8f;

#define N_TOK  4096
#define F_DIM  64
#define B_DIM  4
#define NTILES (N_TOK / 16)   // 256 column tiles
#define NPAIRS (NTILES / 2)   // 128 32-column steps
#define LSTR   80             // P/prob LDS row stride (160B: 16B-aligned, bank-friendly)
#define XSTR   40             // X-tile LDS row stride (80B: 16B-aligned)
#define WAVES  8
#define ROWS_BLK (WAVES * 16) // 128 rows per block

__device__ __forceinline__ v8f wmma_bf16(v16bf a, v16bf b, v8f c) {
  // (neg_a, A, neg_b, B, c_mod, C, reuse_a, reuse_b)
  return __builtin_amdgcn_wmma_f32_16x16x32_bf16(false, a, false, b, (short)0, c, false, false);
}

// A fragment (16x32 bf16) from a row-major bf16 buffer.
// lanes 0-15: row=lane, elems[0..7]=K kb+0..7, [8..15]=K kb+16..23
// lanes 16-31: row=lane-16, elems[0..7]=K kb+8..15, [8..15]=K kb+24..31
template <typename T>
__device__ __forceinline__ v16bf load_afrag(const T* base, int stride, int lane, int kb) {
  int row  = lane & 15;
  int asel = (lane >> 4) * 8;
  const __bf16* p = (const __bf16*)base + row * stride + kb + asel;
  v16bf f;
  ((v8bf*)&f)[0] = *(const v8bf*)(p);
  ((v8bf*)&f)[1] = *(const v8bf*)(p + 16);
  return f;
}

// B fragment (32x16 bf16): B[k][n] = base[n*stride + kb + k]
// lanes 0-15: col n=lane, elems[0..7]=K kb+0..7, [8..15]=K kb+16..23; lanes 16-31: col n=lane-16, elems[0..7]=K kb+8..15, [8..15]=K kb+24..31
template <typename T>
__device__ __forceinline__ v16bf load_bfrag(const T* base, int stride, int lane, int kb) {
  int n  = lane & 15;
  int bk = (lane >> 4) * 8;
  const __bf16* p = (const __bf16*)base + n * stride + kb + bk;
  v16bf f;
  ((v8bf*)&f)[0] = *(const v8bf*)(p);
  ((v8bf*)&f)[1] = *(const v8bf*)(p + 16);
  return f;
}

// ---------------- kernel 0: weight conversion + x transpose + maxnorm init ----------------
__global__ void sgnn_prep(const float* __restrict__ W1, const float* __restrict__ W2,
                          const float* __restrict__ W3, const float* __restrict__ x,
                          __bf16* __restrict__ W1bf, __bf16* __restrict__ W2bf,
                          __bf16* __restrict__ W3bf, __bf16* __restrict__ XbT,
                          unsigned* __restrict__ maxn) {
  int tid = blockIdx.x * blockDim.x + threadIdx.x;
  if (tid < B_DIM) maxn[tid] = 0u;                       // positive-float atomicMax base
  if (tid < F_DIM * F_DIM) {
    W1bf[tid] = (__bf16)W1[tid];
    W2bf[tid] = (__bf16)W2[tid];
    W3bf[tid] = (__bf16)W3[tid];
  }
  const size_t total = (size_t)B_DIM * N_TOK * F_DIM;
  for (size_t idx = tid; idx < total; idx += (size_t)gridDim.x * blockDim.x) {
    size_t b   = idx >> 18;           // N_TOK*F_DIM = 2^18
    size_t rem = idx & ((1u << 18) - 1);
    size_t m   = rem >> 6;
    size_t f   = rem & 63;
    XbT[(b * F_DIM + f) * N_TOK + m] = (__bf16)x[idx];   // [B][64][N] bf16
  }
}

// ---------------- kernel 1: P = (x W1^T + b1) W2^T + b2 -> bf16 ----------------
__global__ __launch_bounds__(32) void sgnn_ff(
    const float* __restrict__ x,
    const __bf16* __restrict__ W1bf, const float* __restrict__ b1,
    const __bf16* __restrict__ W2bf, const float* __restrict__ b2,
    __bf16* __restrict__ Pbf) {
  __shared__ __align__(16) __bf16 t0[16 * LSTR];
  __shared__ __align__(16) __bf16 t1[16 * LSTR];
  const int lane  = threadIdx.x;
  const int row0  = blockIdx.x * 16;            // global row in [0, B*N)
  const int crow0 = (lane >> 4) * 8;            // C-layout row base for this lane
  const int ccol  = lane & 15;                  // C-layout column for this lane

  for (int idx = lane; idx < 16 * F_DIM; idx += 32) {
    int r = idx >> 6, c = idx & 63;
    t0[r * LSTR + c] = (__bf16)x[(size_t)(row0 + r) * F_DIM + c];
  }
  __syncthreads();

  v16bf ax0 = load_afrag(t0, LSTR, lane, 0);
  v16bf ax1 = load_afrag(t0, LSTR, lane, 32);
#pragma unroll
  for (int ft = 0; ft < 4; ++ft) {              // h = x @ W1^T + b1  (B col f = W1 row f)
    v8f acc = {};
    acc = wmma_bf16(ax0, load_bfrag(W1bf + ft * 16 * F_DIM, F_DIM, lane, 0),  acc);
    acc = wmma_bf16(ax1, load_bfrag(W1bf + ft * 16 * F_DIM, F_DIM, lane, 32), acc);
    float bias = b1[ft * 16 + ccol];
#pragma unroll
    for (int r = 0; r < 8; ++r)
      t1[(crow0 + r) * LSTR + ft * 16 + ccol] = (__bf16)(acc[r] + bias);
  }
  __syncthreads();

  v16bf ah0 = load_afrag(t1, LSTR, lane, 0);
  v16bf ah1 = load_afrag(t1, LSTR, lane, 32);
#pragma unroll
  for (int ft = 0; ft < 4; ++ft) {              // P = h @ W2^T + b2
    v8f acc = {};
    acc = wmma_bf16(ah0, load_bfrag(W2bf + ft * 16 * F_DIM, F_DIM, lane, 0),  acc);
    acc = wmma_bf16(ah1, load_bfrag(W2bf + ft * 16 * F_DIM, F_DIM, lane, 32), acc);
    float bias = b2[ft * 16 + ccol];
#pragma unroll
    for (int r = 0; r < 8; ++r)
      Pbf[(size_t)(row0 + crow0 + r) * F_DIM + ft * 16 + ccol] = (__bf16)(acc[r] + bias);
  }
}

// ---------------- kernel 2: row norms of P + per-batch max norm ----------------
// Cauchy-Schwarz: s_nm = p_n . p_m <= |p_n| * max_m |p_m| gives a safe softmax shift
// (shift-invariant; exp(s-M) <= 1, and f32 range keeps full relative precision).
__global__ void sgnn_norms(const __bf16* __restrict__ Pbf,
                           float* __restrict__ rn, unsigned* __restrict__ maxn) {
  int row = blockIdx.x * blockDim.x + threadIdx.x;
  if (row >= B_DIM * N_TOK) return;
  const __bf16* p = Pbf + (size_t)row * F_DIM;
  float s = 0.0f;
#pragma unroll
  for (int c = 0; c < F_DIM; ++c) {
    float v = (float)p[c];
    s += v * v;
  }
  float nrm = sqrtf(s);
  rn[row] = nrm;
  atomicMax(&maxn[row >> 12], __float_as_uint(nrm));   // 4096 rows per batch
}

// ------- kernel 3: fused softmax(P P^T) graph-conv + out-proj -------
// 8 waves/block; shared operands staged once per 32-column step through LDS
// (double-buffered), giving 8x reuse of P / XbT tiles vs per-wave streaming.
// Row sums of R = softmax + I are exactly 2 => D^{-1/2} R D^{-1/2} = R/2.
__global__ __launch_bounds__(256) void sgnn_attn(
    const __bf16* __restrict__ Pbf,   // [B*N][64] bf16
    const __bf16* __restrict__ XbT,   // [B][64][N] bf16 (x transposed)
    const float*  __restrict__ x,     // [B][N][64] f32
    const float*  __restrict__ rn,    // [B*N] row norms of P
    const unsigned* __restrict__ maxn,// [B] max row norm (uint-ordered f32)
    const __bf16* __restrict__ W3bf,  // [64][64] bf16
    const float*  __restrict__ b3,    // [64]
    float* __restrict__ out) {        // [B][N][64] f32
  __shared__ __align__(16) __bf16 PT[2][32 * LSTR];        // staged P column rows   (2x5KB)
  __shared__ __align__(16) __bf16 XT[2][F_DIM * XSTR];     // staged XbT columns     (2x5KB)
  __shared__ __align__(16) __bf16 PR[WAVES][16 * LSTR];    // per-wave prob/out tile (8x2.5KB)

  const int tid   = threadIdx.x;
  const int lane  = tid & 31;
  const int w     = tid >> 5;
  const int b     = blockIdx.x >> 5;            // 32 blocks per batch
  const int it    = blockIdx.x & 31;
  const int row0  = b * N_TOK + it * ROWS_BLK + w * 16;  // wave's global row base
  const int crow0 = (lane >> 4) * 8;
  const int ccol  = lane & 15;

  const __bf16* Pb = Pbf + (size_t)b * N_TOK * F_DIM;
  const __bf16* Xb = XbT + (size_t)b * F_DIM * N_TOK;

  // staging assignment: one b128 per thread per buffer
  const int prow = tid >> 3;         // 0..31
  const int pcol = (tid & 7) * 8;    // 0..56
  const int xrow = tid >> 2;         // 0..63  (feature)
  const int xcol = (tid & 3) * 8;    // 0..24  (column within 32)

  // wave-private A fragments of P rows (K=64)
  v16bf aP0 = load_afrag(Pbf + (size_t)row0 * F_DIM, F_DIM, lane, 0);
  v16bf aP1 = load_afrag(Pbf + (size_t)row0 * F_DIM, F_DIM, lane, 32);

  // analytic per-row softmax shift M_r = |p_r| * max|p| (upper bound of row max)
  const float mnb = __uint_as_float(maxn[b]);
  float mx[8], sm[8];
#pragma unroll
  for (int r = 0; r < 8; ++r) {
    mx[r] = rn[row0 + crow0 + r] * mnb;
    sm[r] = 0.0f;
  }

  // prologue: stage columns [0, 32)
  *(v8bf*)&PT[0][prow * LSTR + pcol] = *(const v8bf*)(Pb + (size_t)prow * F_DIM + pcol);
  *(v8bf*)&XT[0][xrow * XSTR + xcol] = *(const v8bf*)(Xb + (size_t)xrow * N_TOK + xcol);
  __syncthreads();

  v8f O0 = {}, O1 = {}, O2 = {}, O3 = {};
  for (int jp = 0; jp < NPAIRS; ++jp) {
    const int cur = jp & 1;
    if (jp + 1 < NPAIRS) {            // stage next step (overlaps this step's WMMAs)
      const int c0 = (jp + 1) * 32;
      v8bf pv = *(const v8bf*)(Pb + (size_t)(c0 + prow) * F_DIM + pcol);
      v8bf xv = *(const v8bf*)(Xb + (size_t)xrow * N_TOK + c0 + xcol);
      *(v8bf*)&PT[1 - cur][prow * LSTR + pcol] = pv;
      *(v8bf*)&XT[1 - cur][xrow * XSTR + xcol] = xv;
    }
    // S tiles for the two 16-column tiles of this step (B-frags from LDS)
    const __bf16* pt = PT[cur];
    v8f s0 = {}, s1 = {};
    s0 = wmma_bf16(aP0, load_bfrag(pt, LSTR, lane, 0),  s0);
    s0 = wmma_bf16(aP1, load_bfrag(pt, LSTR, lane, 32), s0);
    s1 = wmma_bf16(aP0, load_bfrag(pt + 16 * LSTR, LSTR, lane, 0),  s1);
    s1 = wmma_bf16(aP1, load_bfrag(pt + 16 * LSTR, LSTR, lane, 32), s1);
    __bf16* pr = PR[w];
#pragma unroll
    for (int r = 0; r < 8; ++r) {     // C-layout -> wave-private LDS 16x32 bf16 A-tile
      float e0 = __expf(s0[r] - mx[r]);
      float e1 = __expf(s1[r] - mx[r]);
      sm[r] += e0 + e1;
      pr[(crow0 + r) * LSTR + ccol]      = (__bf16)e0;
      pr[(crow0 + r) * LSTR + 16 + ccol] = (__bf16)e1;
    }
    // same-wave DS ops are in-order: A-frag load sees the stores above
    v16bf ap = load_afrag(pr, LSTR, lane, 0);
    const __bf16* xt = XT[cur];
    O0 = wmma_bf16(ap, load_bfrag(xt +  0 * XSTR, XSTR, lane, 0), O0);
    O1 = wmma_bf16(ap, load_bfrag(xt + 16 * XSTR, XSTR, lane, 0), O1);
    O2 = wmma_bf16(ap, load_bfrag(xt + 32 * XSTR, XSTR, lane, 0), O2);
    O3 = wmma_bf16(ap, load_bfrag(xt + 48 * XSTR, XSTR, lane, 0), O3);
    __syncthreads();                  // staged buffer ready / current buffer released
  }

  // row sums: reduce partials across the 16 lanes of each half-wave
#pragma unroll
  for (int m = 1; m < 16; m <<= 1)
#pragma unroll
    for (int r = 0; r < 8; ++r) sm[r] += __shfl_xor(sm[r], m, 32);

  float scl[8];
#pragma unroll
  for (int r = 0; r < 8; ++r) scl[r] = 0.5f / sm[r];  // fold d_n*d_m = 1/2 with 1/l

  // ---- epilogue: out_core = scl*O + 0.5*x_row -> wave-private LDS for final GEMM ----
  const float* xr = x + (size_t)row0 * F_DIM;
  __bf16* pr = PR[w];
#pragma unroll
  for (int r = 0; r < 8; ++r) {
    int gr = crow0 + r;
    pr[gr * LSTR +  0 + ccol] = (__bf16)(O0[r] * scl[r] + 0.5f * xr[(size_t)gr * F_DIM +  0 + ccol]);
    pr[gr * LSTR + 16 + ccol] = (__bf16)(O1[r] * scl[r] + 0.5f * xr[(size_t)gr * F_DIM + 16 + ccol]);
    pr[gr * LSTR + 32 + ccol] = (__bf16)(O2[r] * scl[r] + 0.5f * xr[(size_t)gr * F_DIM + 32 + ccol]);
    pr[gr * LSTR + 48 + ccol] = (__bf16)(O3[r] * scl[r] + 0.5f * xr[(size_t)gr * F_DIM + 48 + ccol]);
  }
  // wave-private region + in-order DS: no block barrier needed
  v16bf a30 = load_afrag(pr, LSTR, lane, 0);
  v16bf a31 = load_afrag(pr, LSTR, lane, 32);
#pragma unroll
  for (int ft = 0; ft < 4; ++ft) {              // out = relu(core @ W3^T + b3)
    v8f acc = {};
    acc = wmma_bf16(a30, load_bfrag(W3bf + ft * 16 * F_DIM, F_DIM, lane, 0),  acc);
    acc = wmma_bf16(a31, load_bfrag(W3bf + ft * 16 * F_DIM, F_DIM, lane, 32), acc);
    float bias = b3[ft * 16 + ccol];
#pragma unroll
    for (int r = 0; r < 8; ++r) {
      float v = acc[r] + bias;
      out[(size_t)(row0 + crow0 + r) * F_DIM + ft * 16 + ccol] = v > 0.0f ? v : 0.0f;
    }
  }
}

extern "C" void kernel_launch(void* const* d_in, const int* in_sizes, int n_in,
                              void* d_out, int out_size, void* d_ws, size_t ws_size,
                              hipStream_t stream) {
  const float* x  = (const float*)d_in[0];
  const float* W1 = (const float*)d_in[1];
  const float* b1 = (const float*)d_in[2];
  const float* W2 = (const float*)d_in[3];
  const float* b2 = (const float*)d_in[4];
  const float* W3 = (const float*)d_in[5];
  const float* b3 = (const float*)d_in[6];
  float* out = (float*)d_out;

  // workspace: P 2MB | XbT 2MB | W1/2/3 bf16 24KB | rn 64KB | maxn 16B
  char* ws = (char*)d_ws;
  __bf16*   Pbf  = (__bf16*)(ws);
  __bf16*   XbT  = (__bf16*)(ws + ((size_t)2 << 20));
  __bf16*   W1bf = (__bf16*)(ws + ((size_t)4 << 20));
  __bf16*   W2bf = W1bf + F_DIM * F_DIM;
  __bf16*   W3bf = W2bf + F_DIM * F_DIM;
  float*    rn   = (float*)(ws + ((size_t)4 << 20) + (32u << 10));
  unsigned* maxn = (unsigned*)(ws + ((size_t)4 << 20) + (128u << 10));

  sgnn_prep<<<256, 256, 0, stream>>>(W1, W2, W3, x, W1bf, W2bf, W3bf, XbT, maxn);
  sgnn_ff<<<(B_DIM * N_TOK) / 16, 32, 0, stream>>>(x, W1bf, b1, W2bf, b2, Pbf);
  sgnn_norms<<<(B_DIM * N_TOK) / 256, 256, 0, stream>>>(Pbf, rn, maxn);
  sgnn_attn<<<(B_DIM * N_TOK) / ROWS_BLK, WAVES * 32, 0, stream>>>(Pbf, XbT, x, rn, maxn, W3bf, b3, out);
}
